// Router_32006096290574
// MI455X (gfx1250) — matrix operand order repaired and checked by reading, not compiled, not run on hardware
//
#include <hip/hip_runtime.h>

// Router: logits = x @ W^T (bf16 WMMA, f32 acc), top-2 + softmax.
// x: [32768, 2048] f32, W: [64, 2048] f32
// out = weights[32768,2] f32 | indices[32768,2] i32 | logits[32768,64] f32

#define D_DIM   2048
#define N_EXP   64
#define TOKENS  32768

typedef __attribute__((ext_vector_type(16))) __bf16 v16bf;
typedef __attribute__((ext_vector_type(8)))  __bf16 v8bf;
typedef __attribute__((ext_vector_type(4)))  __bf16 v4bf;
typedef __attribute__((ext_vector_type(8)))  float  v8f;
typedef __attribute__((ext_vector_type(4)))  float  v4f;
typedef __attribute__((ext_vector_type(4)))  unsigned int u32x4;
typedef __attribute__((ext_vector_type(8)))  unsigned int u32x8;

// LDS layout produced by TDM with pad_enable: 1 DWORD pad after every 256
// DWORDs (1024 B). Row = 4096 B data + 4x4 B pads = 4112 B pitch.
// Element (row, col): byte = row*4112 + col*2 + (col>>9)*4.
// Bank view: lane lr base = lr*4112 = lr*16 (mod 256) -> each 16-lane half
// covers all 64 banks conflict-free on ds_load_b128.
#define W_ROW_BYTES 4112
#define W_LDS_BYTES (N_EXP * W_ROW_BYTES)   // 263168 B

// ---- Kernel 0: one-shot W f32 -> bf16 into workspace ----
__global__ __launch_bounds__(256)
void convert_w_kernel(const float* __restrict__ W, __bf16* __restrict__ Wbf) {
    const int i = (blockIdx.x * 256 + threadIdx.x) * 4;   // 128 blocks cover 131072 elems
    v4f f = *(const v4f*)(W + i);
    v4bf b;
    b[0] = (__bf16)f[0]; b[1] = (__bf16)f[1];
    b[2] = (__bf16)f[2]; b[3] = (__bf16)f[3];
    *(v4bf*)(Wbf + i) = b;
}

// ---- Kernel 1: WMMA matmul, logits out ----
__global__ __launch_bounds__(512)
void router_mm_kernel(const float* __restrict__ x,
                      const __bf16* __restrict__ Wbf,
                      float* __restrict__ logits) {
    extern __shared__ char smem[];

    const int tid  = threadIdx.x;
    const int wave = tid >> 5;
    const int lane = tid & 31;
    const int half = lane >> 4;   // 0: lanes 0-15, 1: lanes 16-31
    const int lr   = lane & 15;

    // ---- Phase 1: TDM load of bf16 W [64 x 2048] into padded LDS ----
    if (wave == 0) {
        unsigned long long ga = (unsigned long long)(uintptr_t)Wbf;
        u32x4 g0;
        g0[0] = 1u;                                    // count=1 (valid descriptor)
        g0[1] = (unsigned)(uintptr_t)smem;             // lds_addr (byte offset)
        g0[2] = (unsigned)(ga & 0xFFFFFFFFu);          // global_addr[31:0]
        g0[3] = (unsigned)((ga >> 32) & 0x1FFFFFFu)    // global_addr[56:32]
              | (2u << 30);                            // type=2 ("image")
        u32x8 g1;
        g1[0] = 0x01D10000u;  // data_size=1(2B) | pad_enable | pad_interval=7(256 DW)
        g1[1] = 0x08000000u;  // tensor_dim0[15:0]=2048 in [31:16]
        g1[2] = 0x00400000u;  // tensor_dim1[15:0]=64   in [31:16]
        g1[3] = 0x08000000u;  // tile_dim0=2048         in [31:16]
        g1[4] = 0x00000040u;  // tile_dim1=64
        g1[5] = 0x00000800u;  // tensor_dim0_stride=2048 (elements)
        g1[6] = 0u;
        g1[7] = 0u;
        asm volatile("tensor_load_to_lds %0, %1" :: "s"(g0), "s"(g1) : "memory");
        __builtin_amdgcn_s_wait_tensorcnt(0);
    }
    __syncthreads();

    // ---- Phase 2: WMMA main loop. Wave owns 16 tokens x 64 experts. ----
    const int m0 = blockIdx.x * 256 + wave * 16;
    const float* xrow = x + (size_t)(m0 + lr) * D_DIM;

    // Branch-free prefetch clamp: for the very last 16-token tile the +4KB
    // look-ahead would leave x, so retarget its prefetches at the (cached)
    // start of x instead of guarding with a per-iteration divergent branch.
    const float* pfrow = (m0 + 16 < TOKENS) ? xrow : x;

    // Per-N-tile LDS base pointers (include lane row + half offset) so all
    // ds_loads in the unrolled body use base+immediate (no per-iter VALU, no
    // WMMA->VALU WAR hazard NOPs).
    const char* bp0 = smem + lr * W_ROW_BYTES + half * 16;
    const char* bp1 = bp0 + 16 * W_ROW_BYTES;
    const char* bp2 = bp0 + 32 * W_ROW_BYTES;
    const char* bp3 = bp0 + 48 * W_ROW_BYTES;

    v8f acc0 = {}, acc1 = {}, acc2 = {}, acc3 = {};

    for (int seg = 0; seg < 4; ++seg) {      // 4 pad-segments of 512 K-elements
        const float* xs  = xrow  + seg * 512 + half * 8;
        // prefetch base: + half*24 puts the two half-waves on the two distinct
        // 64B lines of each row's 128B per-step window; +1024 = ~32 steps ahead
        const float* pfs = pfrow + seg * 512 + half * 24 + 1024;
        const int sb = seg * 1028;           // 1024 B data + 4 B pad per segment

#pragma unroll
        for (int j = 0; j < 16; ++j) {       // K = 32 per step
            const int ko = j * 32;
            // A fragment (16-bit A 16x32): lane M=lr;
            // elems 0..7 <- K=k0+half*8, elems 8..15 <- K=k0+16+half*8
            v4f a0 = *(const v4f*)(xs + ko);
            v4f a1 = *(const v4f*)(xs + ko + 4);
            v4f a2 = *(const v4f*)(xs + ko + 16);
            v4f a3 = *(const v4f*)(xs + ko + 20);

            // WGP-scope prefetch (scope field omitted = SCOPE_WGP -> fills all
            // cache levels) keeps the HBM stream deep without loadcnt limits.
            asm volatile("global_prefetch_b8 %0, off" :: "v"(pfs + ko));

            v16bf afrag;
#pragma unroll
            for (int i = 0; i < 4; ++i) {
                afrag[i]      = (__bf16)a0[i];
                afrag[4 + i]  = (__bf16)a1[i];
                afrag[8 + i]  = (__bf16)a2[i];
                afrag[12 + i] = (__bf16)a3[i];
            }

            // B fragments (bf16 32x16): lane N = nt*16 + lr,
            // elems 0..7 <- K = k0 + half*8, elems 8..15 <- K = k0 + 16 + half*8
            const int bo = sb + j * 64;      // compile-time immediate per j
            v8bf p0 = *(const v8bf*)(bp0 + bo);
            v8bf p1 = *(const v8bf*)(bp0 + bo + 32);
            v8bf p2 = *(const v8bf*)(bp1 + bo);
            v8bf p3 = *(const v8bf*)(bp1 + bo + 32);
            v8bf p4 = *(const v8bf*)(bp2 + bo);
            v8bf p5 = *(const v8bf*)(bp2 + bo + 32);
            v8bf p6 = *(const v8bf*)(bp3 + bo);
            v8bf p7 = *(const v8bf*)(bp3 + bo + 32);

            v16bf bf0 = __builtin_shufflevector(p0, p1, 0,1,2,3,4,5,6,7,8,9,10,11,12,13,14,15);
            v16bf bf1 = __builtin_shufflevector(p2, p3, 0,1,2,3,4,5,6,7,8,9,10,11,12,13,14,15);
            v16bf bf2 = __builtin_shufflevector(p4, p5, 0,1,2,3,4,5,6,7,8,9,10,11,12,13,14,15);
            v16bf bf3 = __builtin_shufflevector(p6, p7, 0,1,2,3,4,5,6,7,8,9,10,11,12,13,14,15);

            acc0 = __builtin_amdgcn_wmma_f32_16x16x32_bf16(false, afrag, false, bf0,
                                                           (short)0, acc0, false, false);
            acc1 = __builtin_amdgcn_wmma_f32_16x16x32_bf16(false, afrag, false, bf1,
                                                           (short)0, acc1, false, false);
            acc2 = __builtin_amdgcn_wmma_f32_16x16x32_bf16(false, afrag, false, bf2,
                                                           (short)0, acc2, false, false);
            acc3 = __builtin_amdgcn_wmma_f32_16x16x32_bf16(false, afrag, false, bf3,
                                                           (short)0, acc3, false, false);
        }
    }

    // ---- Phase 3: store logits (streamed once -> non-temporal).
    // C/D layout: VGPR r -> M = r + half*8, N = nt*16 + lr.
#pragma unroll
    for (int r = 0; r < 8; ++r) {
        const int tok = m0 + half * 8 + r;
        float* gl = logits + (size_t)tok * N_EXP;
        __builtin_nontemporal_store(acc0[r], gl + lr);
        __builtin_nontemporal_store(acc1[r], gl + 16 + lr);
        __builtin_nontemporal_store(acc2[r], gl + 32 + lr);
        __builtin_nontemporal_store(acc3[r], gl + 48 + lr);
    }
}

// ---- Kernel 2: top-2 + softmax over 64 experts per token ----
__global__ __launch_bounds__(256)
void router_topk_kernel(const float* __restrict__ logits,
                        float* __restrict__ weights,
                        int* __restrict__ indices) {
    __shared__ float Lg[128 * 65];   // 128 tokens x 64 (+1 pad) logits

    const int tid = threadIdx.x;
    const size_t base = (size_t)blockIdx.x * 128 * N_EXP;

    // coalesced staging of 128x64 logits into LDS
#pragma unroll
    for (int i = 0; i < (128 * N_EXP) / 256; ++i) {
        int e   = i * 256 + tid;
        int row = e >> 6;
        int col = e & 63;
        Lg[row * 65 + col] = __builtin_nontemporal_load(logits + base + e);
    }
    __syncthreads();

    if (tid < 128) {
        const float* row = Lg + tid * 65;
        float best = row[0];  int bi = 0;
        float second = -__builtin_inff(); int si = 0;
#pragma unroll 8
        for (int n = 1; n < N_EXP; ++n) {
            float v = row[n];
            if (v > best)        { second = best; si = bi; best = v; bi = n; }
            else if (v > second) { second = v; si = n; }
        }
        const int tok = blockIdx.x * 128 + tid;
        const float w0 = 1.0f / (1.0f + __expf(second - best)); // softmax over top-2
        weights[tok * 2 + 0] = w0;
        weights[tok * 2 + 1] = 1.0f - w0;
        indices[tok * 2 + 0] = bi;
        indices[tok * 2 + 1] = si;
    }
}

extern "C" void kernel_launch(void* const* d_in, const int* in_sizes, int n_in,
                              void* d_out, int out_size, void* d_ws, size_t ws_size,
                              hipStream_t stream) {
    const float* x = (const float*)d_in[0];
    const float* W = (const float*)d_in[1];

    float* out     = (float*)d_out;
    float* weights = out;                       // 32768*2 f32
    int*   indices = (int*)(out + TOKENS * 2);  // 32768*2 i32
    float* logits  = out + TOKENS * 4;          // 32768*64 f32

    __bf16* Wbf = (__bf16*)d_ws;                // 256 KB scratch

    (void)in_sizes; (void)n_in; (void)out_size; (void)ws_size;

    hipFuncSetAttribute((const void*)router_mm_kernel,
                        hipFuncAttributeMaxDynamicSharedMemorySize, W_LDS_BYTES);

    convert_w_kernel<<<dim3((N_EXP * D_DIM) / (256 * 4)), dim3(256), 0, stream>>>(W, Wbf);
    router_mm_kernel<<<dim3(TOKENS / 256), dim3(512), W_LDS_BYTES, stream>>>(x, Wbf, logits);
    router_topk_kernel<<<dim3(TOKENS / 128), dim3(256), 0, stream>>>(logits, weights, indices);
}
